// DenseGATLayer_49563922596601
// MI455X (gfx1250) — matrix operand order repaired and checked by reading, not compiled, not run on hardware
//
#include <hip/hip_runtime.h>
#include <hip/hip_bf16.h>

// ---------------------------------------------------------------------------
// Dense GAT layer on CDNA5 (gfx1250), wave32 + v_wmma_f32_16x16x32_f16.
// B=8, N=1024, Din=256, H=4, F=64 (HF=256).
// ---------------------------------------------------------------------------

#define NNODE 1024
#define DIN   256
#define HF    256
#define NHEAD 4
#define FDIM  64

typedef __attribute__((ext_vector_type(16))) _Float16 v16h;
typedef __attribute__((ext_vector_type(8)))  _Float16 v8h;
typedef __attribute__((ext_vector_type(8)))  float    v8f;
typedef __attribute__((ext_vector_type(4)))  float    f4;
typedef __attribute__((ext_vector_type(4)))  int      i4;

// ---------------------------------------------------------------------------
// Kernel 0: convert W [Din, HF] fp32  ->  Wt [HF, Din] f16 (column-major W)
// ---------------------------------------------------------------------------
__global__ __launch_bounds__(256) void convert_w(const float* __restrict__ W,
                                                 _Float16* __restrict__ Wt) {
    int id   = blockIdx.x * 256 + threadIdx.x;   // 65536 total
    int fcol = id >> 8;                          // 0..255 (output col of W)
    int k    = id & 255;                         // 0..255 (Din index)
    Wt[fcol * DIN + k] = (_Float16)W[k * HF + fcol];
}

// ---------------------------------------------------------------------------
// Kernel 1: Wh = X @ W  via WMMA.  One wave per 16-row strip of X.
// Output WhT stored f-major: WhT[b][f_global][n]  (f16) so the attention
// kernel's B-fragments (K = j) are contiguous per lane.
// ---------------------------------------------------------------------------
__global__ __launch_bounds__(128) void proj_gemm(const float* __restrict__ X,
                                                 const _Float16* __restrict__ Wt,
                                                 _Float16* __restrict__ WhT) {
    const int wave  = threadIdx.x >> 5;
    const int lane  = threadIdx.x & 31;
    const int lrow  = lane & 15;
    const int lhalf = lane >> 4;

    const int strip = blockIdx.x * 4 + wave;     // 0..511
    const int row0  = strip * 16;                // global row base (b*N + n)
    const int b0    = row0 >> 10;
    const int n0    = row0 & 1023;
    const int row   = row0 + lrow;

    const float* __restrict__ xrow = X + (size_t)row * DIN;

    // Cache the whole 16x256 A strip as 8 f16 WMMA A-fragments.
    // A k order used here (contiguous, not AMD's table order): lanes 0-15 hold K = c*32+0..15,
    // lanes 16-31 hold K = c*32+16..31 (matches B fragment order).
    v16h afrag[8];
#pragma unroll
    for (int c = 0; c < 8; ++c) {
        const int k0 = c * 32 + lhalf * 16;
        f4 x0 = *(const f4*)(xrow + k0);
        f4 x1 = *(const f4*)(xrow + k0 + 4);
        f4 x2 = *(const f4*)(xrow + k0 + 8);
        f4 x3 = *(const f4*)(xrow + k0 + 12);
        v16h a;
#pragma unroll
        for (int q = 0; q < 4; ++q) {
            a[q]      = (_Float16)x0[q];
            a[4 + q]  = (_Float16)x1[q];
            a[8 + q]  = (_Float16)x2[q];
            a[12 + q] = (_Float16)x3[q];
        }
        afrag[c] = a;
    }

    // 16 output column tiles of 16 (HF = 256)
#pragma unroll 4
    for (int t = 0; t < 16; ++t) {
        const int fcol = t * 16 + lrow;          // this lane's output column
        const _Float16* __restrict__ wcol = Wt + (size_t)fcol * DIN + lhalf * 16;

        v8f acc = {0.f, 0.f, 0.f, 0.f, 0.f, 0.f, 0.f, 0.f};
#pragma unroll
        for (int c = 0; c < 8; ++c) {
            // B k order used here (same contiguous order as A): lane n holds column n; lanes 0-15 K=c*32+0..15,
            // lanes 16-31 K=c*32+16..31 -> contiguous 16 halves from Wt.
            v16h bfrag = *(const v16h*)(wcol + c * 32);
            acc = __builtin_amdgcn_wmma_f32_16x16x32_f16(
                false, afrag[c], false, bfrag, (short)0, acc, false, false);
        }

        // C layout: lane holds column fcol; VGPR r = row (lhalf*8 + r).
        // Rows are consecutive n indices -> one contiguous 16B f16 store.
        v8h o;
#pragma unroll
        for (int r = 0; r < 8; ++r) o[r] = (_Float16)acc[r];
        *(v8h*)(WhT + ((size_t)b0 * HF + fcol) * NNODE + n0 + lhalf * 8) = o;
    }
}

// ---------------------------------------------------------------------------
// Kernel 2: e_src[b,h,n] = sum_f Wh[b,n,h,f]*a_src[h,f]   (and e_dst)
// ---------------------------------------------------------------------------
__global__ __launch_bounds__(256) void edge_scores(const _Float16* __restrict__ WhT,
                                                   const float* __restrict__ a_src,
                                                   const float* __restrict__ a_dst,
                                                   float* __restrict__ es,
                                                   float* __restrict__ ed) {
    int id = blockIdx.x * 256 + threadIdx.x;     // B*H*N = 32768
    int n  = id & 1023;
    int bh = id >> 10;
    int h  = bh & 3;
    int b  = bh >> 2;
    const _Float16* __restrict__ base = WhT + ((size_t)b * HF + h * FDIM) * NNODE + n;
    float s = 0.f, d = 0.f;
#pragma unroll 8
    for (int f = 0; f < FDIM; ++f) {
        float w = (float)base[(size_t)f * NNODE];
        s = fmaf(w, a_src[h * FDIM + f], s);
        d = fmaf(w, a_dst[h * FDIM + f], d);
    }
    es[id] = s;
    ed[id] = d;
}

// ---------------------------------------------------------------------------
// Kernel 3: masked softmax over j + (alpha @ Wh) via WMMA, + ELU.
// grid: 256 blocks = (b, 32-row i-strip). block: 128 threads = 4 waves = heads.
// Each wave processes TWO 16-row i-tiles so every WMMA B-fragment (head/jt/f
// dependent only) is reused for two WMMAs -> halves WhT L2 traffic.
// The 32x1024 adjacency strip is shared by all heads: compressed ONCE into a
// 4KB LDS bit-matrix via wave32 ballot.  e_dst rows are staged in LDS, so the
// hot loops touch global memory only for B-fragments.
// Pass 1 uses leaky-ReLU monotonicity: row max = lrelu(esrc + masked max of
// e_dst) -> no exp.  Pass 2 builds unnormalized exp(e-m) alpha tiles directly
// in WMMA A-fragment layout; 1/sum folded in the epilogue (LDS transpose).
// ---------------------------------------------------------------------------
__global__ __launch_bounds__(128) void gat_aggregate(const int* __restrict__ Aadj,
                                                     const _Float16* __restrict__ WhT,
                                                     const float* __restrict__ es,
                                                     const float* __restrict__ ed,
                                                     float* __restrict__ out) {
    __shared__ unsigned mbits[32][32];        // bit (j-jt) of word [r][jt>>5]
    __shared__ float    lds_ed[NHEAD][NNODE]; // 16KB: e_dst row per head
    __shared__ float    sinv[NHEAD][32];

    const int h     = threadIdx.x >> 5;          // wave == head
    const int lane  = threadIdx.x & 31;
    const int lrow  = lane & 15;
    const int lhalf = lane >> 4;
    const int o8    = lhalf * 8;

    const int it = blockIdx.x & 31;
    const int b  = blockIdx.x >> 5;
    const int i0 = it * 32;                      // 32-row strip

    const float* __restrict__ esb = es + ((b * NHEAD + h) << 10);
    const float  esrc0 = esb[i0 + lrow];
    const float  esrc1 = esb[i0 + 16 + lrow];
    const float* __restrict__ edrow = ed + ((b * NHEAD + h) << 10);
    const _Float16* __restrict__ bbase = WhT + ((size_t)b * HF + h * FDIM) * NNODE;

    // ---- prologue: stage e_dst row (this head) + build shared bit-mask ----
#pragma unroll
    for (int j = lane * 4; j < NNODE; j += 128)
        *(f4*)&lds_ed[h][j] = *(const f4*)(edrow + j);

    const int* __restrict__ abase = Aadj + ((size_t)b * NNODE + i0) * NNODE;
#pragma unroll 4
    for (int idx = h; idx < 1024; idx += NHEAD) {    // 32 rows x 32 words
        const int r = idx >> 5, c = idx & 31;
        const int a = abase[(size_t)r * NNODE + c * 32 + lane];
        const unsigned bal = __builtin_amdgcn_ballot_w32(a > 0);
        if (lane == 0) mbits[r][c] = bal;
    }
    __syncthreads();

    // ---- pass 1: masked max of e_dst per row (no exp; lrelu is monotone) --
    float dmax0 = -3.0e38f, dmax1 = -3.0e38f;
    for (int jt = 0; jt < NNODE; jt += 32) {
        const unsigned w0 = mbits[lrow][jt >> 5];
        const unsigned w1 = mbits[16 + lrow][jt >> 5];
        const unsigned wa0 = w0 >> o8,        wa1 = w1 >> o8;
        const unsigned wb0 = w0 >> (16 + o8), wb1 = w1 >> (16 + o8);
        const float* ep = &lds_ed[h][jt];
        f4 D0 = *(const f4*)(ep + o8);
        f4 D1 = *(const f4*)(ep + o8 + 4);
        f4 D2 = *(const f4*)(ep + 16 + o8);
        f4 D3 = *(const f4*)(ep + 16 + o8 + 4);
#pragma unroll
        for (int q = 0; q < 4; ++q) {
            dmax0 = fmaxf(dmax0, ((wa0 >> q)       & 1u) ? D0[q] : -3.0e38f);
            dmax0 = fmaxf(dmax0, ((wa0 >> (4 + q)) & 1u) ? D1[q] : -3.0e38f);
            dmax0 = fmaxf(dmax0, ((wb0 >> q)       & 1u) ? D2[q] : -3.0e38f);
            dmax0 = fmaxf(dmax0, ((wb0 >> (4 + q)) & 1u) ? D3[q] : -3.0e38f);
            dmax1 = fmaxf(dmax1, ((wa1 >> q)       & 1u) ? D0[q] : -3.0e38f);
            dmax1 = fmaxf(dmax1, ((wa1 >> (4 + q)) & 1u) ? D1[q] : -3.0e38f);
            dmax1 = fmaxf(dmax1, ((wb1 >> q)       & 1u) ? D2[q] : -3.0e38f);
            dmax1 = fmaxf(dmax1, ((wb1 >> (4 + q)) & 1u) ? D3[q] : -3.0e38f);
        }
    }
    dmax0 = fmaxf(dmax0, __shfl_xor(dmax0, 16, 32)); // combine lane halves
    dmax1 = fmaxf(dmax1, __shfl_xor(dmax1, 16, 32));
    const float xm0 = esrc0 + dmax0, xm1 = esrc1 + dmax1;
    const float mr0 = fmaxf(fmaxf(xm0, 0.2f * xm0), -1.0e9f);
    const float mr1 = fmaxf(fmaxf(xm1, 0.2f * xm1), -1.0e9f);

    // ---- pass 2: unnormalized alpha tiles -> WMMA (B reused x2) ----------
    v8f acc0[4], acc1[4];
#pragma unroll
    for (int t = 0; t < 4; ++t) {
        acc0[t] = (v8f){0.f, 0.f, 0.f, 0.f, 0.f, 0.f, 0.f, 0.f};
        acc1[t] = (v8f){0.f, 0.f, 0.f, 0.f, 0.f, 0.f, 0.f, 0.f};
    }

    float srow0 = 0.f, srow1 = 0.f;
    for (int jt = 0; jt < NNODE; jt += 32) {
        __builtin_prefetch(bbase + (size_t)lrow * NNODE + jt + 64, 0, 0);
        const unsigned w0 = mbits[lrow][jt >> 5];
        const unsigned w1 = mbits[16 + lrow][jt >> 5];
        const float* ep = &lds_ed[h][jt];
        f4 D0 = *(const f4*)(ep + lhalf * 16);
        f4 D1 = *(const f4*)(ep + lhalf * 16 + 4);
        f4 D2 = *(const f4*)(ep + lhalf * 16 + 8);
        f4 D3 = *(const f4*)(ep + lhalf * 16 + 12);

        // p = keep ? exp(lrelu(esrc+ed) - m) : 0, in A-fragment layout
        auto build_af = [&](unsigned w, float esrc, float mr, float& srow) -> v16h {
            const unsigned wa = w >> (lhalf * 16);
            const unsigned wb = w >> (lhalf * 16 + 8);
            v16h af;
#pragma unroll
            for (int q = 0; q < 4; ++q) {
                float x, p;
                x = esrc + D0[q]; x = fmaxf(x, 0.2f * x);
                p = ((wa >> q) & 1u) ? __expf(x - mr) : 0.f;
                srow += p; af[q] = (_Float16)p;
                x = esrc + D1[q]; x = fmaxf(x, 0.2f * x);
                p = ((wa >> (4 + q)) & 1u) ? __expf(x - mr) : 0.f;
                srow += p; af[4 + q] = (_Float16)p;
                x = esrc + D2[q]; x = fmaxf(x, 0.2f * x);
                p = ((wb >> q) & 1u) ? __expf(x - mr) : 0.f;
                srow += p; af[8 + q] = (_Float16)p;
                x = esrc + D3[q]; x = fmaxf(x, 0.2f * x);
                p = ((wb >> (4 + q)) & 1u) ? __expf(x - mr) : 0.f;
                srow += p; af[12 + q] = (_Float16)p;
            }
            return af;
        };
        v16h af0 = build_af(w0, esrc0, mr0, srow0);
        v16h af1 = build_af(w1, esrc1, mr1, srow1);

#pragma unroll
        for (int t = 0; t < 4; ++t) {
            // B-fragment: Wh column f = t*16 + lrow, K = j contiguous in WhT.
            const _Float16* bp = bbase + (size_t)(t * 16 + lrow) * NNODE
                                       + jt + lhalf * 16;
            v16h bf = *(const v16h*)bp;
            acc0[t] = __builtin_amdgcn_wmma_f32_16x16x32_f16(
                false, af0, false, bf, (short)0, acc0[t], false, false);
            acc1[t] = __builtin_amdgcn_wmma_f32_16x16x32_f16(
                false, af1, false, bf, (short)0, acc1[t], false, false);
        }
    }

    // total row sums (both halves), transpose lane->VGPR via LDS
    float st0 = srow0 + __shfl_xor(srow0, 16, 32);
    float st1 = srow1 + __shfl_xor(srow1, 16, 32);
    sinv[h][lrow]      = 1.f / fmaxf(st0, 1.0e-30f);
    sinv[h][16 + lrow] = 1.f / fmaxf(st1, 1.0e-30f);
    __syncthreads();

    float si0[8], si1[8];
#pragma unroll
    for (int r = 0; r < 8; ++r) {
        si0[r] = sinv[h][o8 + r];
        si1[r] = sinv[h][16 + o8 + r];
    }

    // ---- epilogue: normalize, ELU, store fp32 out[b][i][h*64+f] ----
    float* __restrict__ ob0 = out + ((size_t)b * NNODE + i0) * HF + h * FDIM;
    float* __restrict__ ob1 = ob0 + (size_t)16 * HF;
#pragma unroll
    for (int t = 0; t < 4; ++t) {
        const int fc = t * 16 + lrow;
#pragma unroll
        for (int r = 0; r < 8; ++r) {
            float x = acc0[t][r] * si0[r];
            x = (x > 0.f) ? x : (__expf(x) - 1.f);          // ELU
            ob0[(size_t)(o8 + r) * HF + fc] = x;
            float y = acc1[t][r] * si1[r];
            y = (y > 0.f) ? y : (__expf(y) - 1.f);
            ob1[(size_t)(o8 + r) * HF + fc] = y;
        }
    }
}

// ---------------------------------------------------------------------------
// Launch
// ---------------------------------------------------------------------------
extern "C" void kernel_launch(void* const* d_in, const int* in_sizes, int n_in,
                              void* d_out, int out_size, void* d_ws, size_t ws_size,
                              hipStream_t stream) {
    const float* X     = (const float*)d_in[0];   // [8,1024,256]
    const int*   A     = (const int*)d_in[1];     // [8,1024,1024]
    const float* W     = (const float*)d_in[2];   // [256,256]
    const float* a_src = (const float*)d_in[3];   // [4,64]
    const float* a_dst = (const float*)d_in[4];   // [4,64]
    float*       out   = (float*)d_out;           // [8,1024,256]

    char* ws = (char*)d_ws;
    _Float16* Wt  = (_Float16*)ws;                                   // 128 KB
    _Float16* WhT = (_Float16*)(ws + 131072);                        // 4 MB
    float*    es  = (float*)(ws + 131072 + 4194304);                 // 128 KB
    float*    ed  = (float*)(ws + 131072 + 4194304 + 131072);        // 128 KB

    convert_w    <<<256, 256, 0, stream>>>(W, Wt);
    proj_gemm    <<<128, 128, 0, stream>>>(X, Wt, WhT);
    edge_scores  <<<128, 256, 0, stream>>>(WhT, a_src, a_dst, es, ed);
    gat_aggregate<<<256, 128, 0, stream>>>(A, WhT, es, ed, out);
}
